// TripletLoss_16990890623003
// MI455X (gfx1250) — compile-verified, matrix operand order checked by reading, not run on hardware
//
#include <hip/hip_runtime.h>
#include <hip/hip_bf16.h>

#define N_PTS 8192
#define DIM   2048
#define NCLS  512
#define BT    128          // block tile (rows and cols)
#define KT    32           // bf16 WMMA K
#define NTB   (N_PTS / BT) // 64 tiles per side
#define LDA   40           // padded LDS stride in halves (80B rows -> conflict-free b128 reads)
#define INF_BITS 0x7149F2CAu  // bits of 1e30f

typedef __attribute__((ext_vector_type(16))) __bf16 v16bf;
typedef __attribute__((ext_vector_type(8)))  __bf16 v8bf;
typedef __attribute__((ext_vector_type(8)))  float  v8f;
typedef __attribute__((ext_vector_type(4)))  unsigned int u32x4;
typedef __attribute__((ext_vector_type(8)))  int i32x8;
typedef __attribute__((ext_vector_type(4)))  int i32x4;

static __device__ inline unsigned short f2bf_rne(float x) {
    unsigned u = __float_as_uint(x);
    unsigned r = u + 0x7FFFu + ((u >> 16) & 1u);   // round-to-nearest-even
    return (unsigned short)(r >> 16);
}

static __device__ inline float bf2f(unsigned short h) {
    return __uint_as_float(((unsigned)h) << 16);
}

static __device__ inline unsigned lds_off(const void* p) {
    // LDS aperture: addr[31:0] is the LDS byte offset (ISA 10.2 aperture table)
    return (unsigned)(unsigned long long)p;
}

// Issue one TDM DMA: 128 rows x 32 bf16 from global (row stride DIM) into LDS,
// padding 4 DWORDs after every 16 DWORDs so rows land at the LDA=40-half stride.
// D# layout per CDNA5 ISA.
static __device__ inline void tdm_load_tile(const unsigned short* gbase, unsigned ldsOff) {
    unsigned long long ga = (unsigned long long)gbase;
    u32x4 g0 = { 1u,                                   // count=1, is_restore=0, gather=0
                 ldsOff,                               // lds_addr (bytes)
                 (unsigned)(ga & 0xFFFFFFFFull),       // global_addr[31:0]
                 (unsigned)((ga >> 32) & 0x01FFFFFFull) | (2u << 30) }; // addr[56:32] | type=2
    i32x8 g1 = { (int)((1u << 16)    // data_size = 2 bytes
                 | (1u << 20)        // pad_enable
                 | (3u << 22)        // pad_interval: 16 DWORDs (64B row)
                 | (3u << 25)),      // pad_amount: 4 DWORDs (16B)
                 (int)(((unsigned)DIM) << 16),     // tensor_dim0[15:0]=2048 @ bits 63:48
                 (int)(((unsigned)N_PTS) << 16),   // dim0 hi=0, tensor_dim1[15:0]=8192
                 (int)(32u << 16),                 // dim1 hi=0, tile_dim0=32
                 128,                              // tile_dim1=128, tile_dim2=0
                 DIM,                              // tensor_dim0_stride = 2048
                 0, 0 };                           // stride hi, tensor_dim1_stride=0
    i32x4 z4 = { 0, 0, 0, 0 };
    i32x8 z8 = { 0, 0, 0, 0, 0, 0, 0, 0 };
    // 6-arg form on this toolchain: (g0, g1, g2, g3, <aux group>, cpol)
    __builtin_amdgcn_tensor_load_to_lds(g0, g1, z4, z4, z8, 0);
}

// Build a 16x32 bf16 WMMA fragment for this lane from an LDS row.
// Layout per ISA 7.12.2: halves[0..7] = K k0..k0+7, halves[8..15] = K k0+16..k0+23,
// with k0 = (lane<16 ? 0 : 8).
static __device__ inline v16bf make_frag(const unsigned short* p, int k0) {
    v8bf a = *(const v8bf*)(p + k0);
    v8bf b = *(const v8bf*)(p + k0 + 16);
    v16bf r;
#pragma unroll
    for (int i = 0; i < 8; ++i) { r[i] = a[i]; r[i + 8] = b[i]; }
    return r;
}

// ---------------- kernel 1: row sum-of-squares + bf16 hi/lo split ----------------
__global__ __launch_bounds__(256)
void k_prep(const float* __restrict__ X,
            unsigned short* __restrict__ Xhi,
            unsigned short* __restrict__ Xlo,
            float* __restrict__ sq) {
    const int row = blockIdx.x;
    const int tid = threadIdx.x;
    const float* xr = X + (size_t)row * DIM;

    float4 v0 = ((const float4*)xr)[tid * 2 + 0];
    float4 v1 = ((const float4*)xr)[tid * 2 + 1];
    float e[8] = { v0.x, v0.y, v0.z, v0.w, v1.x, v1.y, v1.z, v1.w };

    union { unsigned short us[8]; uint4 v; } ph, pl;
    float ss = 0.f;
#pragma unroll
    for (int i = 0; i < 8; ++i) {
        float x = e[i];
        ss += x * x;
        unsigned short h = f2bf_rne(x);
        float res = x - bf2f(h);
        ph.us[i] = h;
        pl.us[i] = f2bf_rne(res);
    }
    ((uint4*)(Xhi + (size_t)row * DIM))[tid] = ph.v;
    ((uint4*)(Xlo + (size_t)row * DIM))[tid] = pl.v;

    __shared__ float red[256];
    red[tid] = ss;
    __syncthreads();
    for (int s = 128; s > 0; s >>= 1) {
        if (tid < s) red[tid] += red[tid + s];
        __syncthreads();
    }
    if (tid == 0) sq[row] = red[0];
}

// ---------------- kernel 2: init reduction arrays + class histogram ----------------
__global__ void k_init(unsigned* __restrict__ apb, unsigned* __restrict__ anb,
                       int* __restrict__ hist) {
    int g = blockIdx.x * 256 + threadIdx.x;
    apb[g] = 0u;
    anb[g] = INF_BITS;
    if (g < NCLS) hist[g] = 0;
}

__global__ void k_hist(const int* __restrict__ tg, int* __restrict__ hist) {
    int g = blockIdx.x * 256 + threadIdx.x;
    atomicAdd(&hist[tg[g]], 1);
}

// ---------------- kernel 3: fused WMMA GEMM + distance + hard-mining reduce ----------------
__global__ __launch_bounds__(256)
void k_gemm_dist(const unsigned short* __restrict__ Xhi,
                 const unsigned short* __restrict__ Xlo,
                 const float* __restrict__ sq,
                 const int* __restrict__ tg,
                 unsigned* __restrict__ apb,
                 unsigned* __restrict__ anb) {
    // double-buffered TDM-filled tiles: [buffer][hi/lo][128 rows * LDA halves]
    __shared__ unsigned short sA[2][2][BT * LDA];
    __shared__ unsigned short sB[2][2][BT * LDA];
    __shared__ float    s_sqr[BT], s_sqc[BT];
    __shared__ int      s_clr[BT], s_clc[BT];
    __shared__ unsigned s_apr[BT], s_anr[BT], s_apc[BT], s_anc[BT];

    // decode upper-triangular tile index -> (br, bc), bc >= br
    int rem = blockIdx.x, br = 0;
    while (rem >= (NTB - br)) { rem -= (NTB - br); ++br; }
    const int bc = br + rem;

    const int tid = threadIdx.x;
    if (tid < BT) {
        s_apr[tid] = 0u;       s_anr[tid] = INF_BITS;
        s_apc[tid] = 0u;       s_anc[tid] = INF_BITS;
        s_sqr[tid] = sq[br * BT + tid];  s_clr[tid] = tg[br * BT + tid];
        s_sqc[tid] = sq[bc * BT + tid];  s_clc[tid] = tg[bc * BT + tid];
    }

    const int wave = tid >> 5, lane = tid & 31;
    const int wm = wave >> 2, wn = wave & 3;     // 2x4 waves -> 64x32 per wave
    const int half = lane >> 4, lr = lane & 15;
    const int k0 = half * 8;

    const size_t aBase = (size_t)(br * BT) * DIM;
    const size_t bBase = (size_t)(bc * BT) * DIM;

    v8f acc[4][2];
#pragma unroll
    for (int mt = 0; mt < 4; ++mt)
#pragma unroll
        for (int nt = 0; nt < 2; ++nt)
#pragma unroll
            for (int i = 0; i < 8; ++i) acc[mt][nt][i] = 0.f;

    // prologue: DMA first K-tile into buffer 0 (wave 0 only; TDM is a wave-level op)
    if (wave == 0) {
        tdm_load_tile(Xhi + aBase, lds_off(&sA[0][0][0]));
        tdm_load_tile(Xlo + aBase, lds_off(&sA[0][1][0]));
        tdm_load_tile(Xhi + bBase, lds_off(&sB[0][0][0]));
        tdm_load_tile(Xlo + bBase, lds_off(&sB[0][1][0]));
    }

    for (int kk = 0; kk < DIM; kk += KT) {
        const int p = (kk >> 5) & 1;
        __syncthreads();                       // everyone done reading buffer p^1
        if (wave == 0) {
            const int kn = kk + KT;
            if (kn < DIM) {
                const int q = p ^ 1;
                tdm_load_tile(Xhi + aBase + kn, lds_off(&sA[q][0][0]));
                tdm_load_tile(Xlo + aBase + kn, lds_off(&sA[q][1][0]));
                tdm_load_tile(Xhi + bBase + kn, lds_off(&sB[q][0][0]));
                tdm_load_tile(Xlo + bBase + kn, lds_off(&sB[q][1][0]));
                // TDM completes in order: <=4 outstanding => buffer p's 4 ops done
                __builtin_amdgcn_s_wait_tensorcnt(4);
            } else {
                __builtin_amdgcn_s_wait_tensorcnt(0);
            }
        }
        __syncthreads();                       // buffer p ready for all waves

        const unsigned short* Ah = sA[p][0];
        const unsigned short* Al = sA[p][1];
        const unsigned short* Bh = sB[p][0];
        const unsigned short* Bl = sB[p][1];

        v16bf afh[4], afl[4], bfh[2], bfl[2];
#pragma unroll
        for (int mt = 0; mt < 4; ++mt) {
            const int r = (wm * 64 + mt * 16 + lr) * LDA;
            afh[mt] = make_frag(&Ah[r], k0);
            afl[mt] = make_frag(&Al[r], k0);
        }
#pragma unroll
        for (int nt = 0; nt < 2; ++nt) {
            const int r = (wn * 32 + nt * 16 + lr) * LDA;
            bfh[nt] = make_frag(&Bh[r], k0);
            bfl[nt] = make_frag(&Bl[r], k0);
        }
#pragma unroll
        for (int mt = 0; mt < 4; ++mt)
#pragma unroll
            for (int nt = 0; nt < 2; ++nt) {
                v8f c = acc[mt][nt];
                // dot ~= hi*hi + hi*lo + lo*hi  (fp32-accurate bf16x3 split)
                c = __builtin_amdgcn_wmma_f32_16x16x32_bf16(
                        false, afh[mt], false, bfh[nt], (short)0, c, false, false);
                c = __builtin_amdgcn_wmma_f32_16x16x32_bf16(
                        false, afh[mt], false, bfl[nt], (short)0, c, false, false);
                c = __builtin_amdgcn_wmma_f32_16x16x32_bf16(
                        false, afl[mt], false, bfh[nt], (short)0, c, false, false);
                acc[mt][nt] = c;
            }
    }

    // ---- epilogue: distances + masked max/min, symmetric (row & col anchors) ----
    __syncthreads();
#pragma unroll
    for (int mt = 0; mt < 4; ++mt)
#pragma unroll
        for (int nt = 0; nt < 2; ++nt)
#pragma unroll
            for (int e = 0; e < 8; ++e) {
                const int rloc = wm * 64 + mt * 16 + e + 8 * half; // C layout: M = e + 8*(lane>=16)
                const int cloc = wn * 32 + nt * 16 + lr;           // N = lane&15
                const float dot = acc[mt][nt][e];
                float d2 = s_sqr[rloc] + s_sqc[cloc] - 2.0f * dot;
                float d  = sqrtf(fmaxf(d2, 1e-12f));
                unsigned db = __float_as_uint(d);  // monotone for d >= 0
                if (s_clr[rloc] == s_clc[cloc]) {
                    atomicMax(&s_apr[rloc], db);
                    atomicMax(&s_apc[cloc], db);
                } else {
                    atomicMin(&s_anr[rloc], db);
                    atomicMin(&s_anc[cloc], db);
                }
            }
    __syncthreads();
    if (tid < BT) {
        atomicMax(&apb[br * BT + tid], s_apr[tid]);
        atomicMin(&anb[br * BT + tid], s_anr[tid]);
        atomicMax(&apb[bc * BT + tid], s_apc[tid]);
        atomicMin(&anb[bc * BT + tid], s_anc[tid]);
    }
}

// ---------------- kernel 4: final margin-ranking reduction ----------------
__global__ __launch_bounds__(256)
void k_final(const int* __restrict__ tg,
             const unsigned* __restrict__ apb,
             const unsigned* __restrict__ anb,
             const int* __restrict__ hist,
             float* __restrict__ out) {
    __shared__ float red[256];
    const int tid = threadIdx.x;
    float local = 0.f;
    for (int i = tid; i < N_PTS; i += 256) {
        const int pc = hist[tg[i]];
        const int nc = N_PTS - pc;
        float dap = (pc > 1) ? __uint_as_float(apb[i]) : 0.0f;
        float dan = (nc > 0) ? __uint_as_float(anb[i]) : 1.0e6f;
        local += fmaxf(dap - dan + 0.3f, 0.0f);
    }
    red[tid] = local;
    __syncthreads();
    for (int s = 128; s > 0; s >>= 1) {
        if (tid < s) red[tid] += red[tid + s];
        __syncthreads();
    }
    if (tid == 0) out[0] = red[0] / (float)N_PTS;
}

// ---------------- host launcher ----------------
extern "C" void kernel_launch(void* const* d_in, const int* in_sizes, int n_in,
                              void* d_out, int out_size, void* d_ws, size_t ws_size,
                              hipStream_t stream) {
    const float* X  = (const float*)d_in[0];
    const int*   tg = (const int*)d_in[1];
    float* out = (float*)d_out;

    char* ws = (char*)d_ws;
    size_t off = 0;
    unsigned short* Xhi = (unsigned short*)(ws + off); off += (size_t)N_PTS * DIM * 2;
    unsigned short* Xlo = (unsigned short*)(ws + off); off += (size_t)N_PTS * DIM * 2;
    float*    sq   = (float*)(ws + off);    off += (size_t)N_PTS * 4;
    unsigned* apb  = (unsigned*)(ws + off); off += (size_t)N_PTS * 4;
    unsigned* anb  = (unsigned*)(ws + off); off += (size_t)N_PTS * 4;
    int*      hist = (int*)(ws + off);      off += (size_t)NCLS * 4;

    k_prep<<<N_PTS, 256, 0, stream>>>(X, Xhi, Xlo, sq);
    k_init<<<N_PTS / 256, 256, 0, stream>>>(apb, anb, hist);
    k_hist<<<N_PTS / 256, 256, 0, stream>>>(tg, hist);
    k_gemm_dist<<<NTB * (NTB + 1) / 2, 256, 0, stream>>>(Xhi, Xlo, sq, tg, apb, anb);
    k_final<<<1, 256, 0, stream>>>(tg, apb, anb, hist, out);
}
